// denoise_Pool_v1_52278341927164
// MI455X (gfx1250) — matrix operand order repaired and checked by reading, not compiled, not run on hardware
//
#include <hip/hip_runtime.h>
#include <hip/hip_bf16.h>

typedef __attribute__((ext_vector_type(16))) _Float16 v16h;
typedef __attribute__((ext_vector_type(8)))  float    v8f;
typedef __attribute__((ext_vector_type(4)))  unsigned int v4u;
typedef __attribute__((ext_vector_type(8)))  int      v8i;
typedef __attribute__((ext_vector_type(4)))  int      v4i;

#define NODES 8192
#define KNBR  64
#define FD    256
#define HDIM  128
#define NEDGE (NODES*KNBR)      /* 524288 */
#define NKEEP 32

/* output tuple offsets (flat f32 concat in return order) */
#define O_ADJ   0L
#define O_FEAT  (O_ADJ  + (long)NODES*NODES)   /* 67108864 */
#define O_PROB  (O_FEAT + (long)NODES*FD)      /* 69206016 */
#define O_PAIR  (O_PROB + (long)NEDGE*2)       /* 70254592 */
#define O_BEF   (O_PAIR + (long)NEDGE*2)       /* 71303168 */
#define O_AFT   (O_BEF + 1)                    /* 71303169 */
#define O_LEFT  (O_AFT + 1)                    /* 71303170 */

#if defined(__AMDGCN__) && __has_builtin(__builtin_amdgcn_tensor_load_to_lds) && \
    __has_builtin(__builtin_amdgcn_s_wait_tensorcnt)
#define USE_TDM 1
#endif

/* ---- scalar init: before_edge_num = K, zero after-edge accumulator ---- */
__global__ void init_scalars_kernel(float* s) {
    s[0] = (float)KNBR;   /* before_edge_num */
    s[1] = 0.0f;          /* after_edge_num accumulator */
}

/* ---- convert W1 (f32 [256][128]) into WMMA B-fragment-ordered f16 ----
   B-matrix (32x16 f16) layout: lane L holds column N = nn*16 + L%16, element t is
   K = kk*32 + (t<8 ? 8h+t : 16+8h+t-8), h = L/16.  Fragment index = (((kk*8+nn)*32+lane)*16+t). */
__global__ void prep_w1_kernel(const float* __restrict__ W1, _Float16* __restrict__ frag) {
    int idx  = blockIdx.x * 256 + threadIdx.x;   /* 32768 entries */
    int t    = idx & 15;
    int lane = (idx >> 4) & 31;
    int nn   = (idx >> 9) & 7;
    int kk   = idx >> 12;
    int kr   = (t < 8) ? (8 * (lane >> 4) + t) : (16 + 8 * (lane >> 4) + (t - 8));
    frag[idx] = (_Float16)W1[(kk * 32 + kr) * HDIM + nn * 16 + (lane & 15)];
}

/* ---- edge MLP: diff@W1 (+b1, PReLU) folded head -> sigmoid keep-prob ----
   Block = 256 threads = 8 waves; wave = 16-edge M tile.  64 WMMAs/wave.
   W1 fragments staged to LDS by the Tensor Data Mover (wave 0 issues the DMA,
   other waves overlap their global feat-diff loads with the transfer). */
__global__ void __launch_bounds__(256)
edge_mlp_kernel(const float* __restrict__ feat, const int* __restrict__ nbr,
                const _Float16* __restrict__ w1frag,
                const float* __restrict__ b1, const float* __restrict__ pa,
                const float* __restrict__ W2, const float* __restrict__ b2,
                float* __restrict__ prob_out, float* __restrict__ p1_ws) {
    __shared__ __align__(32) _Float16 sB[8 * 8 * 32 * 16];   /* 64 KB */

    int tid  = threadIdx.x;
    int wave = tid >> 5;
    int lane = tid & 31;

#ifdef USE_TDM
    if (wave == 0) {
        /* Tensor DMA: 1 row x 8192 8-byte elements, global -> LDS.
           D# group0: count=1 | lds_addr | 57b global_addr | type=2.      */
        unsigned long long ga = (unsigned long long)w1frag;
        unsigned           la = (unsigned)(unsigned long long)(const void*)&sB[0];
        v4u g0 = { 1u,                                   /* count=1 */
                   la,                                   /* lds_addr */
                   (unsigned)(ga & 0xFFFFFFFFu),
                   (unsigned)((ga >> 32) & 0x01FFFFFFu) | 0x80000000u /* type=2 */ };
        v8i g1 = { (int)(3u << 16),          /* data_size=8B, no mask/pad/iter */
                   (int)((8192u & 0xFFFFu) << 16),   /* tensor_dim0 lo16 */
                   (int)(1u << 16),          /* tensor_dim0 hi16=0 | tensor_dim1=1 */
                   (int)(8192u << 16),       /* tensor_dim1 hi16=0 | tile_dim0=8192 */
                   1,                        /* tile_dim1=1 | tile_dim2=0 */
                   8192,                     /* tensor_dim0_stride lo32 */
                   0, 0 };
        v4i g2 = { 0, 0, 0, 0 };
        v4i g3 = { 0, 0, 0, 0 };
#if defined(__clang_major__) && (__clang_major__ >= 23)
        v8i g4 = { 0, 0, 0, 0, 0, 0, 0, 0 };
        __builtin_amdgcn_tensor_load_to_lds(g0, g1, g2, g3, g4, 0);
#else
        __builtin_amdgcn_tensor_load_to_lds(g0, g1, g2, g3, 0);
#endif
        __builtin_amdgcn_s_wait_tensorcnt(0);
    }
#else
    /* fallback: coalesced manual stage of pre-converted fragments */
    {
        const uint4* srcp = (const uint4*)w1frag;
        uint4*       dstp = (uint4*)sB;
        #pragma unroll
        for (int i = 0; i < 16; ++i) dstp[tid + 256 * i] = srcp[tid + 256 * i];
    }
#endif

    int tileBase = blockIdx.x * 128 + wave * 16;

    /* Build A fragments (16x32 f16 per kk) in registers while TDM runs.
       lane<16: row M=lane, K runs {0..7,16..23}; lane>=16: same M, {8..15,24..31}. */
    int m = lane & 15;
    int e = tileBase + m;
    int src_row = e >> 6;          /* pair_src = repeat(arange(N), K) */
    int dst_row = nbr[e];
    const float* fs = feat + (long)src_row * FD;
    const float* fd = feat + (long)dst_row * FD;
    int kbase = (lane < 16) ? 0 : 8;

    v16h A[8];
    #pragma unroll
    for (int kk = 0; kk < 8; ++kk) {
        int k0 = kk * 32 + kbase;
        #pragma unroll
        for (int j = 0; j < 8; ++j) {
            A[kk][j]     = (_Float16)(fs[k0 + j]      - fd[k0 + j]);
            A[kk][8 + j] = (_Float16)(fs[k0 + 16 + j] - fd[k0 + 16 + j]);
        }
    }

    __syncthreads();   /* publish TDM-filled LDS to all waves */

    v8f acc[8];
    #pragma unroll
    for (int nn = 0; nn < 8; ++nn) acc[nn] = (v8f){0.f,0.f,0.f,0.f,0.f,0.f,0.f,0.f};

    #pragma unroll
    for (int kk = 0; kk < 8; ++kk) {
        #pragma unroll
        for (int nn = 0; nn < 8; ++nn) {
            v16h B = *(const v16h*)&sB[(((kk * 8 + nn) * 32 + lane) << 4)];
            acc[nn] = __builtin_amdgcn_wmma_f32_16x16x32_f16(
                false, A[kk], false, B, (short)0, acc[nn], false, false);
        }
    }

    /* epilogue: +b1, PReLU, dot with (W2[:,1]-W2[:,0]).
       C layout: VGPR r, lanes 0-15 -> (M=r, N=lane); lanes 16-31 -> (M=r+8, N=lane-16). */
    int nloc = lane & 15;
    float partial[8] = {0.f,0.f,0.f,0.f,0.f,0.f,0.f,0.f};
    #pragma unroll
    for (int nn = 0; nn < 8; ++nn) {
        int   n  = nn * 16 + nloc;
        float bb = b1[n], aa = pa[n];
        float wd = W2[n * 2 + 1] - W2[n * 2 + 0];
        #pragma unroll
        for (int r = 0; r < 8; ++r) {
            float h = acc[nn][r] + bb;
            h = (h >= 0.f) ? h : aa * h;
            partial[r] += h * wd;
        }
    }
    /* reduce over the 16 lanes of each half (xor stays within halves) */
    #pragma unroll
    for (int r = 0; r < 8; ++r) {
        float v = partial[r];
        v += __shfl_xor(v, 1, 32);
        v += __shfl_xor(v, 2, 32);
        v += __shfl_xor(v, 4, 32);
        v += __shfl_xor(v, 8, 32);
        partial[r] = v;
    }
    float b2d = b2[1] - b2[0];
    if (lane == 0 || lane == 16) {
        int mbase = (lane == 0) ? 0 : 8;
        #pragma unroll
        for (int r = 0; r < 8; ++r) {
            int   ee = tileBase + mbase + r;
            float z  = partial[r] + b2d;
            float p1 = 1.f / (1.f + __expf(-z));   /* softmax[,1] == sigmoid(l1-l0) */
            p1_ws[ee]            = p1;
            prob_out[ee * 2 + 0] = 1.f - p1;
            prob_out[ee * 2 + 1] = p1;
        }
    }
}

/* ---- per-row: stable rank of keep-prob, drop 32 lowest, degree-normalize ---- */
__global__ void __launch_bounds__(256)
select_norm_kernel(const float* __restrict__ adj, const int* __restrict__ nbr,
                   const float* __restrict__ p1, float* __restrict__ new_adj,
                   float* __restrict__ left_out, float* __restrict__ after_acc) {
    __shared__ float sp[KNBR];
    __shared__ int   sdrop[NKEEP];
    __shared__ float sdeg;
    __shared__ int   scnt;

    int row = blockIdx.x;
    int tid = threadIdx.x;
    if (tid < KNBR) sp[tid] = p1[row * KNBR + tid];
    if (tid == 0) { sdeg = 0.f; scnt = 0; }
    __syncthreads();

    if (tid < KNBR) {
        float me = sp[tid];
        int rank = 0;
        #pragma unroll
        for (int j = 0; j < KNBR; ++j) {
            float o = sp[j];
            rank += (o < me) || (o == me && j < tid);   /* stable ascending */
        }
        int col = nbr[row * KNBR + tid];
        if (rank < KNBR - NKEEP) {
            sdrop[rank] = col;
        } else {
            left_out[row * NKEEP + (rank - (KNBR - NKEEP))] =
                (float)(row * KNBR + tid);
            float v = adj[(long)row * NODES + col];
            atomicAdd(&sdeg, v);
            if (v != 0.f) atomicAdd(&scnt, 1);
        }
    }
    __syncthreads();

    float inv = 1.f / (sdeg + 1e-6f);
    const float4* arow = (const float4*)(adj     + (long)row * NODES);
    float4*       orow = (float4*)(new_adj + (long)row * NODES);
    __builtin_prefetch(arow + tid, 0, 0);
    #pragma unroll
    for (int i = 0; i < 8; ++i) {
        float4 v = arow[tid + 256 * i];
        v.x *= inv; v.y *= inv; v.z *= inv; v.w *= inv;
        orow[tid + 256 * i] = v;
    }
    __threadfence();
    __syncthreads();
    if (tid < NKEEP) new_adj[(long)row * NODES + sdrop[tid]] = 0.f;
    if (tid == 0) atomicAdd(after_acc, (float)scnt);
}

/* ---- feat copy + pair_list (int values stored exactly as f32) ---- */
__global__ void feat_pair_kernel(const float* __restrict__ feat,
                                 const int* __restrict__ nbr,
                                 float* __restrict__ out_feat,
                                 float* __restrict__ out_pair) {
    int tid = blockIdx.x * 256 + threadIdx.x;   /* 2,097,152 threads */
    out_feat[tid] = feat[tid];
    if (tid < NEDGE) {
        out_pair[tid * 2 + 0] = (float)(tid >> 6);
        out_pair[tid * 2 + 1] = (float)nbr[tid];
    }
}

__global__ void finalize_kernel(float* a) { a[0] = a[0] / (float)NODES; }

extern "C" void kernel_launch(void* const* d_in, const int* in_sizes, int n_in,
                              void* d_out, int out_size, void* d_ws, size_t ws_size,
                              hipStream_t stream) {
    const float* adj  = (const float*)d_in[0];
    const float* feat = (const float*)d_in[1];
    const int*   nbr  = (const int*)d_in[2];
    /* d_in[3] = dst_num (compile-time constant here) */
    const float* W1 = (const float*)d_in[4];
    const float* b1 = (const float*)d_in[5];
    const float* pa = (const float*)d_in[6];
    const float* W2 = (const float*)d_in[7];
    const float* b2 = (const float*)d_in[8];
    float* out = (float*)d_out;

    float*    p1_ws = (float*)d_ws;                               /* E f32 = 2 MB */
    _Float16* frag  = (_Float16*)((char*)d_ws + (size_t)NEDGE*4); /* 64 KB */

    init_scalars_kernel<<<1, 1, 0, stream>>>(out + O_BEF);
    prep_w1_kernel<<<128, 256, 0, stream>>>(W1, frag);
    edge_mlp_kernel<<<NEDGE / 128, 256, 0, stream>>>(
        feat, nbr, frag, b1, pa, W2, b2, out + O_PROB, p1_ws);
    select_norm_kernel<<<NODES, 256, 0, stream>>>(
        adj, nbr, p1_ws, out + O_ADJ, out + O_LEFT, out + O_AFT);
    feat_pair_kernel<<<(NODES * FD) / 256, 256, 0, stream>>>(
        feat, nbr, out + O_FEAT, out + O_PAIR);
    finalize_kernel<<<1, 1, 0, stream>>>(out + O_AFT);
}
